// FreqAwareExpertChoiceMoE_51342039056409
// MI455X (gfx1250) — matrix operand order repaired and checked by reading, not compiled, not run on hardware
//
#include <hip/hip_runtime.h>
#include <hip/hip_bf16.h>

// ---- problem constants (match reference) ----
#define KD    256      // input_size
#define KF    64       // dct features
#define KE    8        // experts
#define KNB   4        // bands
#define KH    512      // hidden
#define KDO   256      // output
#define KR    8        // lora rank
#define KN    65536    // tokens
#define KCAP  10240    // capacity per expert
#define KBM   32       // token rows per MoE block
#define KKP1  288      // K for layer1, 257 padded to 9*32
#define KSCALE 2.0f    // alpha / rank = 16/8

typedef __attribute__((ext_vector_type(16))) __bf16 v16bf;
typedef __attribute__((ext_vector_type(8)))  float  v8f;

// ---- order-preserving float <-> uint keys (for max / radix-select) ----
__device__ __forceinline__ unsigned fkey(float f) {
  unsigned u = __float_as_uint(f);
  return (u & 0x80000000u) ? ~u : (u | 0x80000000u);
}
__device__ __forceinline__ float funkey(unsigned k) {
  unsigned u = (k & 0x80000000u) ? (k & 0x7fffffffu) : ~k;
  return __uint_as_float(u);
}

// ---- WMMA fragment loaders (bf16, 16x16x32 shapes) ----
// A: 16 rows (M) x 32 cols (K), row-major; lane r=M, VGPR v -> K=(v<4?0:16)+8*grp+2*(v&3)
//    -> dwords {4*grp+0..3, 8+4*grp+0..3} = uint4 q[grp], q[2+grp]
__device__ __forceinline__ v16bf frag_a(const __bf16* base, int stride, int lane) {
  int grp = (lane >> 4) & 1;
  int r   = lane & 15;
  const uint4* p = (const uint4*)(base + (size_t)r * stride);
  union { v16bf v; uint4 q[2]; } f;
  f.q[0] = p[grp];
  f.q[1] = p[2 + grp];
  return f.v;
}
// B: stored N-major (row = N, 32 K cols); lane r=N, VGPR v -> K=(v<4?0:16)+8*grp+2*(v&3)
//    -> dwords {4*grp+0..3, 8+4*grp+0..3} = uint4 q[grp], q[2+grp]
__device__ __forceinline__ v16bf frag_b(const __bf16* base, int stride, int lane) {
  int grp = (lane >> 4) & 1;
  int r   = lane & 15;
  const uint4* p = (const uint4*)(base + (size_t)r * stride);
  union { v16bf v; uint4 q[2]; } f;
  f.q[0] = p[grp];
  f.q[1] = p[2 + grp];
  return f.v;
}

__device__ __forceinline__ unsigned pack_bf16(float v0, float v1) {
  union { __bf16 h[2]; unsigned u; } pk;
  pk.h[0] = (__bf16)v0;
  pk.h[1] = (__bf16)v1;
  return pk.u;
}

__device__ __forceinline__ float gelu_tanh(float h) {
  float t = 0.7978845608028654f * (h + 0.044715f * h * h * h);
  return 0.5f * h * (1.0f + tanhf(t));
}

// ============ kernel 0: zero output + stats ============
__global__ __launch_bounds__(256) void k_zero(float* __restrict__ y,
                                              unsigned* __restrict__ maxbits,
                                              float* __restrict__ sumexp) {
  size_t i = (size_t)blockIdx.x * 256 + threadIdx.x;
  if (i < (size_t)KN * KDO) y[i] = 0.0f;
  if (blockIdx.x == 0 && threadIdx.x < KE) {
    maxbits[threadIdx.x] = 0u;   // maps to most-negative float
    sumexp[threadIdx.x]  = 0.0f;
  }
}

// ============ kernel 1: fold DCT into gate:  Wg_eff = W_gate[:D] + W_dct @ W_gate[D:] ============
__global__ __launch_bounds__(256) void k_wgeff(const float* __restrict__ W_dct,
                                               const float* __restrict__ W_gate,
                                               float* __restrict__ wgeff) {
  int d = threadIdx.x;  // 0..255
#pragma unroll
  for (int e = 0; e < KE; ++e) {
    float acc = W_gate[d * KE + e];
    for (int f = 0; f < KF; ++f)
      acc += W_dct[d * KF + f] * W_gate[(KD + f) * KE + e];
    wgeff[d * KE + e] = acc;
  }
}

// ============ kernel 1b: pre-transpose + convert expert weights to bf16 ============
// W1T[e][n][kp] (kp in 0..287, zero-padded past 257); W2T[e][n][k] (k in 0..511)
__global__ __launch_bounds__(256) void k_wconv(const float* __restrict__ W1,
                                               const float* __restrict__ W2,
                                               unsigned* __restrict__ W1T,
                                               unsigned* __restrict__ W2T) {
  int gid = blockIdx.x * 256 + threadIdx.x;
  const int n1 = KE * KH * (KKP1 / 2);   // packed dwords in W1T
  if (gid < n1) {
    int kp2  = gid % (KKP1 / 2);
    int rest = gid / (KKP1 / 2);
    int n = rest % KH;
    int e = rest / KH;
    int k0 = kp2 * 2, k1 = k0 + 1;
    float v0 = (k0 < KD + 1) ? W1[((size_t)e * (KD + 1) + k0) * KH + n] : 0.0f;
    float v1 = (k1 < KD + 1) ? W1[((size_t)e * (KD + 1) + k1) * KH + n] : 0.0f;
    W1T[gid] = pack_bf16(v0, v1);
  } else {
    int g = gid - n1;
    const int n2 = KE * KDO * (KH / 2);
    if (g < n2) {
      int kp2  = g % (KH / 2);
      int rest = g / (KH / 2);
      int n = rest % KDO;
      int e = rest / KDO;
      int k0 = kp2 * 2;
      float v0 = W2[((size_t)e * KH + k0) * KDO + n];
      float v1 = W2[((size_t)e * KH + k0 + 1) * KDO + n];
      W2T[g] = pack_bf16(v0, v1);
    }
  }
}

// ============ kernel 2: logits = x @ Wg_eff  (+ per-expert max via atomicMax on keys) ============
__global__ __launch_bounds__(256) void k_logits(const float* __restrict__ x,
                                                const float* __restrict__ wgeff,
                                                float* __restrict__ logits,
                                                unsigned* __restrict__ maxbits) {
  __shared__ float swg[KD * KE];
  __shared__ float red[256];
  int tid = threadIdx.x;
  for (int i = tid; i < KD * KE; i += 256) swg[i] = wgeff[i];
  __syncthreads();
  int n = blockIdx.x * 256 + tid;
  const float* xp = x + (size_t)n * KD;
  float acc[KE];
#pragma unroll
  for (int e = 0; e < KE; ++e) acc[e] = 0.0f;
  for (int d = 0; d < KD; ++d) {
    float xv = xp[d];
#pragma unroll
    for (int e = 0; e < KE; ++e) acc[e] += xv * swg[d * KE + e];
  }
#pragma unroll
  for (int e = 0; e < KE; ++e) logits[(size_t)n * KE + e] = acc[e];
  for (int e = 0; e < KE; ++e) {
    red[tid] = acc[e];
    __syncthreads();
    for (int s = 128; s > 0; s >>= 1) {
      if (tid < s) red[tid] = fmaxf(red[tid], red[tid + s]);
      __syncthreads();
    }
    if (tid == 0) atomicMax(maxbits + e, fkey(red[0]));
    __syncthreads();
  }
}

// ============ kernel 3: per-expert sum of exp(logit - max) ============
__global__ __launch_bounds__(256) void k_sumexp(const float* __restrict__ logits,
                                                const unsigned* __restrict__ maxbits,
                                                float* __restrict__ sumexp) {
  __shared__ float red[256];
  __shared__ float mv[KE];
  int tid = threadIdx.x;
  if (tid < KE) mv[tid] = funkey(maxbits[tid]);
  __syncthreads();
  int n = blockIdx.x * 256 + tid;
  float v[KE];
#pragma unroll
  for (int e = 0; e < KE; ++e) v[e] = expf(logits[(size_t)n * KE + e] - mv[e]);
  for (int e = 0; e < KE; ++e) {
    red[tid] = v[e];
    __syncthreads();
    for (int s = 128; s > 0; s >>= 1) {
      if (tid < s) red[tid] += red[tid + s];
      __syncthreads();
    }
    if (tid == 0) atomicAdd(sumexp + e, red[0]);
    __syncthreads();
  }
}

// ============ kernel 4: exact CAP-th-largest threshold via 4-pass radix select ============
__global__ __launch_bounds__(256) void k_select(const float* __restrict__ logits,
                                                unsigned* __restrict__ thresh,
                                                unsigned* __restrict__ eqneed,
                                                unsigned* __restrict__ cnt,
                                                unsigned* __restrict__ cnteq) {
  __shared__ unsigned hist[256];
  __shared__ unsigned sh_bin, sh_want;
  int e = blockIdx.x, tid = threadIdx.x;
  unsigned prefix = 0, pmask = 0;
  unsigned want = KCAP;
  for (int pass = 0; pass < 4; ++pass) {
    int shift = 24 - 8 * pass;
    hist[tid] = 0;
    __syncthreads();
    for (int n = tid; n < KN; n += 256) {
      unsigned key = fkey(logits[(size_t)n * KE + e]);
      if ((key & pmask) == prefix)
        atomicAdd(&hist[(key >> shift) & 255], 1u);
    }
    __syncthreads();
    if (tid == 0) {
      unsigned cum = 0, b = 0, w = want;
      for (int i = 255; i >= 0; --i) {
        unsigned c = hist[i];
        if (cum + c >= want) { b = (unsigned)i; w = want - cum; break; }
        cum += c;
      }
      sh_bin = b; sh_want = w;
    }
    __syncthreads();
    prefix |= (sh_bin << shift);
    pmask  |= (0xFFu << shift);
    want = sh_want;
    __syncthreads();
  }
  if (tid == 0) {
    thresh[e] = prefix; eqneed[e] = want;
    cnt[e] = 0; cnteq[e] = 0;
  }
}

// ============ kernel 5: compact selected tokens (order irrelevant: final op is scatter-add) ============
__global__ __launch_bounds__(256) void k_compact(const float* __restrict__ logits,
                                                 const int* __restrict__ band,
                                                 const unsigned* __restrict__ maxbits,
                                                 const float* __restrict__ sumexp,
                                                 const unsigned* __restrict__ thresh,
                                                 const unsigned* __restrict__ eqneed,
                                                 unsigned* __restrict__ cnt,
                                                 unsigned* __restrict__ cnteq,
                                                 int* __restrict__ idxbuf,
                                                 int* __restrict__ bandbuf,
                                                 float* __restrict__ gbuf) {
  int n = blockIdx.x * 256 + threadIdx.x;
  int bnd = band[n];
#pragma unroll
  for (int e = 0; e < KE; ++e) {
    float lg = logits[(size_t)n * KE + e];
    unsigned key = fkey(lg);
    unsigned T = thresh[e];
    bool take = false;
    if (key > T) take = true;
    else if (key == T) {
      unsigned q = atomicAdd(&cnteq[e], 1u);
      if (q < eqneed[e]) take = true;
    }
    if (take) {
      unsigned slot = atomicAdd(&cnt[e], 1u);
      if (slot < KCAP) {
        float m = funkey(maxbits[e]);
        float g = expf(lg - m) / sumexp[e];
        idxbuf[e * KCAP + slot]  = n;
        bandbuf[e * KCAP + slot] = bnd;
        gbuf[e * KCAP + slot]    = g;
      }
    }
  }
}

// ============ kernel 6: fused per-expert 2-layer MLP with per-band LoRA, bf16 WMMA ============
// One block = 32 tokens of one expert; 256 threads = 8 waves (2x4 WMMA tile grid per 32x64 tile).
// B-fragments come straight from the pre-transposed bf16 weights (2x global_load_b128/lane);
// the k-loop has no barriers and no LDS weight staging.
__global__ __launch_bounds__(256) void k_moe(const float* __restrict__ x,
                                             const int* __restrict__ idxbuf,
                                             const int* __restrict__ bandbuf,
                                             const float* __restrict__ gbuf,
                                             const __bf16* __restrict__ W1T,
                                             const float* __restrict__ b1,
                                             const float* __restrict__ A1,
                                             const float* __restrict__ B1,
                                             const __bf16* __restrict__ W2T,
                                             const float* __restrict__ b2,
                                             const float* __restrict__ A2,
                                             const float* __restrict__ B2,
                                             float* __restrict__ y) {
  __shared__ __bf16 xls[KBM][KKP1];   // gathered x_full tile, bf16, K padded (18 KB)
  __shared__ __bf16 hls[KBM][KH];     // hidden activations, bf16             (32 KB)
  __shared__ int   toks[KBM];
  __shared__ int   bnds[KBM];
  __shared__ float gv[KBM];
  __shared__ float u1[KBM][KR];       // xe @ A1[band]
  __shared__ float u2[KBM][KR];       // h  @ A2[band]

  const int tid  = threadIdx.x;
  const int e    = blockIdx.x / (KCAP / KBM);
  const int mb   = blockIdx.x % (KCAP / KBM);
  const int slot0 = mb * KBM;
  const int lane = tid & 31;
  const int w    = tid >> 5;
  const int mt   = w >> 2;       // 0..1
  const int nt   = w & 3;        // 0..3
  const int grp  = (lane >> 4) & 1;
  const int lrow = lane & 15;

  // ---- phase 0: token metadata ----
  if (tid < KBM) {
    int s = slot0 + tid;
    toks[tid] = idxbuf[e * KCAP + s];
    bnds[tid] = bandbuf[e * KCAP + s];
    gv[tid]   = gbuf[e * KCAP + s];
  }
  __syncthreads();

  // ---- phase 0b: gather x_full rows into LDS (bf16 pairs, b32 stores) ----
  {
    int m  = tid >> 3;                 // 0..31
    int c0 = (tid & 7) * (KKP1 / 8);   // multiple of 36 (even)
    int tok = toks[m];
    float bf = (float)bnds[m] * (1.0f / KNB);
    const float* xp = x + (size_t)tok * KD;
    unsigned* row = (unsigned*)&xls[m][0];
    for (int c = c0; c < c0 + (KKP1 / 8); c += 2) {
      float v0 = (c     < KD) ? xp[c]     : ((c     == KD) ? bf : 0.0f);
      float v1 = (c + 1 < KD) ? xp[c + 1] : ((c + 1 == KD) ? bf : 0.0f);
      row[c >> 1] = pack_bf16(v0, v1);
    }
  }
  __syncthreads();

  // ---- phase 1: u1 = xe @ A1[band]  (fp32, rank-8) ----
  {
    int m = tid >> 3, r = tid & 7;
    int b = bnds[m];
    const float* A1p = A1 + ((size_t)(e * KNB + b) * (KD + 1)) * KR + r;
    float acc = 0.0f;
    for (int k = 0; k < KD + 1; ++k)
      acc += (float)xls[m][k] * A1p[(size_t)k * KR];
    u1[m][r] = acc;
  }
  __syncthreads();

  // ---- phase 2: layer 1 GEMM (bf16 WMMA, f32 acc) + bias + LoRA + GELU -> hls ----
  for (int nb1 = 0; nb1 < KH / 64; ++nb1) {
    v8f acc = {0.f, 0.f, 0.f, 0.f, 0.f, 0.f, 0.f, 0.f};
    const __bf16* wrow = W1T + ((size_t)e * KH + nb1 * 64 + nt * 16) * KKP1;
#pragma unroll
    for (int ks = 0; ks < KKP1 / 32; ++ks) {
      v16bf af  = frag_a(&xls[mt * 16][ks * 32], KKP1, lane);
      v16bf bfr = frag_b(wrow + ks * 32, KKP1, lane);
      acc = __builtin_amdgcn_wmma_f32_16x16x32_bf16(false, af, false, bfr,
                                                    (short)0, acc, false, false);
    }
    {
      int hcol = nb1 * 64 + nt * 16 + lrow;
      float bias = b1[e * KH + hcol];
#pragma unroll
      for (int j = 0; j < 8; ++j) {
        int m = mt * 16 + grp * 8 + j;
        int b = bnds[m];
        const float* B1p = B1 + ((size_t)(e * KNB + b) * KR) * KH + hcol;
        float lo = 0.0f;
#pragma unroll
        for (int r = 0; r < KR; ++r) lo += u1[m][r] * B1p[(size_t)r * KH];
        float hval = acc[j] + bias + KSCALE * lo;
        hls[m][hcol] = (__bf16)gelu_tanh(hval);
      }
    }
  }
  __syncthreads();

  // ---- phase 3: u2 = h @ A2[band]  (fp32, rank-8) ----
  {
    int m = tid >> 3, r = tid & 7;
    int b = bnds[m];
    const float* A2p = A2 + ((size_t)(e * KNB + b) * KH) * KR + r;
    float acc = 0.0f;
    for (int h = 0; h < KH; ++h)
      acc += (float)hls[m][h] * A2p[(size_t)h * KR];
    u2[m][r] = acc;
  }
  __syncthreads();

  // ---- phase 4: layer 2 GEMM + bias + LoRA + gate-weighted scatter-add ----
  for (int nb2 = 0; nb2 < KDO / 64; ++nb2) {
    v8f acc = {0.f, 0.f, 0.f, 0.f, 0.f, 0.f, 0.f, 0.f};
    const __bf16* wrow = W2T + ((size_t)e * KDO + nb2 * 64 + nt * 16) * KH;
#pragma unroll
    for (int ks = 0; ks < KH / 32; ++ks) {
      v16bf af  = frag_a(&hls[mt * 16][ks * 32], KH, lane);
      v16bf bfr = frag_b(wrow + ks * 32, KH, lane);
      acc = __builtin_amdgcn_wmma_f32_16x16x32_bf16(false, af, false, bfr,
                                                    (short)0, acc, false, false);
    }
    {
      int ocol = nb2 * 64 + nt * 16 + lrow;
      float bias = b2[e * KDO + ocol];
#pragma unroll
      for (int j = 0; j < 8; ++j) {
        int m = mt * 16 + grp * 8 + j;
        int b = bnds[m];
        const float* B2p = B2 + ((size_t)(e * KNB + b) * KR) * KDO + ocol;
        float lo = 0.0f;
#pragma unroll
        for (int r = 0; r < KR; ++r) lo += u2[m][r] * B2p[(size_t)r * KDO];
        float o = acc[j] + bias + KSCALE * lo;
        atomicAdd(&y[(size_t)toks[m] * KDO + ocol], gv[m] * o);
      }
    }
  }
}

// ============ host launcher ============
extern "C" void kernel_launch(void* const* d_in, const int* in_sizes, int n_in,
                              void* d_out, int out_size, void* d_ws, size_t ws_size,
                              hipStream_t stream) {
  const float* x      = (const float*)d_in[0];
  const int*   band   = (const int*)  d_in[1];
  const float* W_dct  = (const float*)d_in[2];
  const float* W_gate = (const float*)d_in[3];
  const float* W1     = (const float*)d_in[4];
  const float* b1     = (const float*)d_in[5];
  const float* A1     = (const float*)d_in[6];
  const float* B1     = (const float*)d_in[7];
  const float* W2     = (const float*)d_in[8];
  const float* b2     = (const float*)d_in[9];
  const float* A2     = (const float*)d_in[10];
  const float* B2     = (const float*)d_in[11];
  float* y = (float*)d_out;

  char* ws = (char*)d_ws;
  size_t o = 0;
  auto take = [&](size_t bytes) -> char* {
    char* p = ws + o;
    o += (bytes + 255) & ~(size_t)255;
    return p;
  };
  float*    logits  = (float*)   take((size_t)KN * KE * 4);
  float*    wgeff   = (float*)   take((size_t)KD * KE * 4);
  unsigned* maxbits = (unsigned*)take(KE * 4);
  float*    sumexp  = (float*)   take(KE * 4);
  unsigned* thresh  = (unsigned*)take(KE * 4);
  unsigned* eqneed  = (unsigned*)take(KE * 4);
  unsigned* cnt     = (unsigned*)take(KE * 4);
  unsigned* cnteq   = (unsigned*)take(KE * 4);
  int*      idxbuf  = (int*)     take((size_t)KE * KCAP * 4);
  int*      bandbuf = (int*)     take((size_t)KE * KCAP * 4);
  float*    gbuf    = (float*)   take((size_t)KE * KCAP * 4);
  unsigned* W1T     = (unsigned*)take((size_t)KE * KH * KKP1 * 2);  // bf16, packed
  unsigned* W2T     = (unsigned*)take((size_t)KE * KDO * KH * 2);   // bf16, packed
  (void)in_sizes; (void)n_in; (void)out_size; (void)ws_size;

  const int wconv_dwords = KE * KH * (KKP1 / 2) + KE * KDO * (KH / 2);

  k_zero   <<<(KN * KDO) / 256, 256, 0, stream>>>(y, maxbits, sumexp);
  k_wgeff  <<<1, 256, 0, stream>>>(W_dct, W_gate, wgeff);
  k_wconv  <<<(wconv_dwords + 255) / 256, 256, 0, stream>>>(W1, W2, W1T, W2T);
  k_logits <<<KN / 256, 256, 0, stream>>>(x, wgeff, logits, maxbits);
  k_sumexp <<<KN / 256, 256, 0, stream>>>(logits, maxbits, sumexp);
  k_select <<<KE, 256, 0, stream>>>(logits, thresh, eqneed, cnt, cnteq);
  k_compact<<<KN / 256, 256, 0, stream>>>(logits, band, maxbits, sumexp,
                                          thresh, eqneed, cnt, cnteq,
                                          idxbuf, bandbuf, gbuf);
  k_moe    <<<KE * (KCAP / KBM), 256, 0, stream>>>(x, idxbuf, bandbuf, gbuf,
                                                   (const __bf16*)W1T, b1, A1, B1,
                                                   (const __bf16*)W2T, b2, A2, B2, y);
}
